// Loss_6279242187481
// MI455X (gfx1250) — matrix operand order repaired and checked by reading, not compiled, not run on hardware
//
#include <hip/hip_runtime.h>
#include <hip/hip_bf16.h>
#include <math.h>

// Problem constants (from reference): B=4096, N=2B=8192, d=128, TEMP=0.5
#define B_SIZE 4096
#define N_SIZE 8192
#define D      128
#define NWAVE  8            // waves per block (wave32)
#define COL_CHUNKS 4        // column split across blocks
#define TILES_PER_CHUNK (N_SIZE / 16 / COL_CHUNKS)   // 128

#define LOG2E_F 1.4426950408889634f
#define LN2_F   0.6931471805599453f
#define K_SCALE (2.0f * LOG2E_F)      // (1/TEMP) * log2(e): base-2 logits
#define NEG_BIG (-3.0e38f)            // finite "-inf" for running max init

// d_ws float layout: pos[N] | mx[N][4] | sm[N][4] | hb (bf16 h, N*D ushorts)
#define HB_OFF_FLOATS (N_SIZE + 2 * N_SIZE * COL_CHUNKS)   // 73728 floats (16B aligned)

typedef __attribute__((ext_vector_type(16))) __bf16 v16bf;
typedef __attribute__((ext_vector_type(8)))  float  v8f;

union FragU {
    uint4 u[2];   // 32 bytes
    v16bf v;      // 16 bf16 = 32 bytes
};

__device__ __forceinline__ unsigned short f32_to_bf16(float f) {
    unsigned int u = __float_as_uint(f);
    unsigned int r = u + 0x7FFFu + ((u >> 16) & 1u);   // round-to-nearest-even
    return (unsigned short)(r >> 16);
}

__device__ __forceinline__ const float* row_ptr(const float* hi, const float* hj, int i) {
    return (i < B_SIZE) ? (hi + (size_t)i * D) : (hj + (size_t)(i - B_SIZE) * D);
}

// ---------------------------------------------------------------------------
// Kernel P: one-time f32 -> bf16 conversion of h = [h_i; h_j] into hb.
// Removes ~67M redundant per-block conversions from the hot GEMM loop.
// ---------------------------------------------------------------------------
__global__ __launch_bounds__(256) void simclr_prep(
    const float* __restrict__ hi, const float* __restrict__ hj,
    unsigned short* __restrict__ hb)
{
    const int t    = blockIdx.x * 256 + threadIdx.x;   // 0 .. N*D/8 - 1
    const int base = t * 8;
    const int row  = base >> 7;
    const int col  = base & (D - 1);
    const float* src = row_ptr(hi, hj, row) + col;
    float4 x = *(const float4*)(src);
    float4 y = *(const float4*)(src + 4);
    uint4 o;
    o.x = (unsigned)f32_to_bf16(x.x) | ((unsigned)f32_to_bf16(x.y) << 16);
    o.y = (unsigned)f32_to_bf16(x.z) | ((unsigned)f32_to_bf16(x.w) << 16);
    o.z = (unsigned)f32_to_bf16(y.x) | ((unsigned)f32_to_bf16(y.y) << 16);
    o.w = (unsigned)f32_to_bf16(y.z) | ((unsigned)f32_to_bf16(y.w) << 16);
    *(uint4*)(hb + base) = o;
}

// ---------------------------------------------------------------------------
// Kernel 0: positives. pos2[n] = (1/T)*log2(e) * dot(h[n], h[(n+B)%N]) in f32.
// ---------------------------------------------------------------------------
__global__ __launch_bounds__(256) void simclr_pos(
    const float* __restrict__ hi, const float* __restrict__ hj,
    float* __restrict__ ws)
{
    const int n = blockIdx.x * 256 + threadIdx.x;        // 0..N-1
    const int m = (n + B_SIZE) & (N_SIZE - 1);
    const float4* a = (const float4*)row_ptr(hi, hj, n);
    const float4* b = (const float4*)row_ptr(hi, hj, m);
    float d = 0.0f;
#pragma unroll
    for (int k = 0; k < D / 4; ++k) {
        float4 x = a[k], y = b[k];
        d = __builtin_fmaf(x.x, y.x, d);
        d = __builtin_fmaf(x.y, y.y, d);
        d = __builtin_fmaf(x.z, y.z, d);
        d = __builtin_fmaf(x.w, y.w, d);
    }
    ws[n] = d * K_SCALE;   // base-2 units
}

// ---------------------------------------------------------------------------
// Kernel 1: fused sim-tile GEMM (bf16 WMMA) + weighting + masking + online LSE
// grid = 256 blocks: blockIdx/4 = row-block (128 rows), blockIdx%4 = col chunk
// block = 256 threads = 8 waves; wave w owns rows [rowblock*128 + w*16, +16)
// B tiles move global->LDS with GLOBAL_LOAD_ASYNC_TO_LDS_B128 (ASYNCcnt),
// double-buffered, overlapped with WMMA + epilogue of the current tile.
// A fragments load directly from global bf16 (16B-contiguous chunks).
// ---------------------------------------------------------------------------
__global__ __launch_bounds__(256) void simclr_rowpass(
    const unsigned short* __restrict__ hb,
    const float* __restrict__ S, float* __restrict__ ws)
{
    __shared__ unsigned short ldsB[2][16 * D];   // double-buffered B tile, col-major bf16
    __shared__ float redM[NWAVE][16][16];
    __shared__ float redS[NWAVE][16][16];

    const int tid = threadIdx.x;
    const int w   = tid >> 5;        // wave id in block
    const int l   = tid & 31;        // lane
    const int blockRow = blockIdx.x >> 2;
    const int chunk    = blockIdx.x & 3;
    // force wave-uniform values into SGPRs so tile-mask checks are scalar
    const int i0 = __builtin_amdgcn_readfirstlane((blockRow * NWAVE + w) * 16);
    const int dTile = i0;                              // diagonal-crossing tile
    const int pTile = (i0 + B_SIZE) & (N_SIZE - 1);    // positive-band tile

    float* mxW = ws + N_SIZE;                          // [N][COL_CHUNKS] base-2 max
    float* smW = ws + N_SIZE + N_SIZE * COL_CHUNKS;    // [N][COL_CHUNKS] base-2 sums

    // per-thread async-copy slice of a B tile: thread copies 16B of column c
    const int cCol  = tid >> 4;          // 0..15 tile column
    const int cPart = tid & 15;          // 16B chunk within the 256B column
    // flat->LDS: low 32 bits of a shared-aperture address are the LDS offset
    const unsigned ldsOff0 = (unsigned)(uintptr_t)(&ldsB[0][0])
                           + (unsigned)(cCol * 256 + cPart * 16);

    // ---- build 4 A fragments (K chunks of 32) per lane, direct from global ----
    // A layout: lanes 0-15: halves 0-7 -> K=base+0..7,  halves 8-15 -> K=base+16..23
    //           lanes16-31: halves 0-7 -> K=base+8..15, halves 8-15 -> K=base+24..31
    const int g  = l >> 4;
    const int mA = l & 15;
    const unsigned short* aRow = hb + (size_t)(i0 + mA) * D;
    FragU afr[4];
#pragma unroll
    for (int kc = 0; kc < 4; ++kc) {
        afr[kc].u[0] = *(const uint4*)(aRow + kc * 32 + 8 * g);
        afr[kc].u[1] = *(const uint4*)(aRow + kc * 32 + 16 + 8 * g);
    }

    // online base-2 logsumexp accumulators: lane l covers col (l&15) of each
    // tile; rows m = g*8 + r (matches the 16x16 f32 C/D VGPR layout)
    float mx[8], sm[8];
    int   iRow[8], jpRow[8];
    size_t srow[8];
#pragma unroll
    for (int r = 0; r < 8; ++r) {
        mx[r]    = NEG_BIG;
        sm[r]    = 0.0f;
        iRow[r]  = i0 + g * 8 + r;
        jpRow[r] = (iRow[r] + B_SIZE) & (N_SIZE - 1);
        srow[r]  = (size_t)(iRow[r] & (B_SIZE - 1)) * B_SIZE;
    }
    const int nCol = l & 15;

    // ---- prologue: async-copy tile 0 into ldsB[0] ----
    {
        const int j0 = chunk * TILES_PER_CHUNK * 16;
        const unsigned short* gsrc = hb + (size_t)(j0 + cCol) * D + cPart * 8;
        asm volatile("global_load_async_to_lds_b128 %0, %1, off"
                     :: "v"(ldsOff0), "v"(gsrc) : "memory");
    }

    for (int t = 0; t < TILES_PER_CHUNK; ++t) {
        const int cur = t & 1;
        const int j0  = (chunk * TILES_PER_CHUNK + t) * 16;

        // our async writes into ldsB[cur] must land before the barrier
        asm volatile("s_wait_asynccnt 0x0" ::: "memory");
        __syncthreads();   // + compiler's dscnt wait retires ldsB[cur^1] readers

        // ---- issue async copy of tile t+1 into ldsB[cur^1] ----
        if (t + 1 < TILES_PER_CHUNK) {
            const int j0n = j0 + 16;
            const unsigned short* gsrc = hb + (size_t)(j0n + cCol) * D + cPart * 8;
            const unsigned ldsOff = ldsOff0 + (unsigned)((cur ^ 1) * 16 * D * 2);
            asm volatile("global_load_async_to_lds_b128 %0, %1, off"
                         :: "v"(ldsOff), "v"(gsrc) : "memory");
            if (t + 2 < TILES_PER_CHUNK) {   // prefetch tile t+2 into near caches
                const unsigned short* p = hb + (size_t)(j0 + 32 + cCol) * D + cPart * 8;
                __builtin_prefetch(p, 0, 3);
            }
        }

        // ---- 16x16 sim tile via 4 bf16 WMMAs over K=128 ----
        v8f acc = {0.f, 0.f, 0.f, 0.f, 0.f, 0.f, 0.f, 0.f};
#pragma unroll
        for (int kc = 0; kc < 4; ++kc) {
            // B layout: same as A (lanes 0-15: K=base+0..7,16..23; lanes 16-31: K=base+8..15,24..31)
            FragU bfr;
            const unsigned short* colp = &ldsB[cur][nCol * D + kc * 32 + 8 * g];
            bfr.u[0] = *(const uint4*)(colp);
            bfr.u[1] = *(const uint4*)(colp + 16);
            acc = __builtin_amdgcn_wmma_f32_16x16x32_bf16(
                false, afr[kc].v, false, bfr.v, (short)0, acc, false, false);
        }

        // ---- epilogue: weight by 1-S^2, online base-2 LSE ----
        const int j    = j0 + nCol;
        const int jMod = j & (B_SIZE - 1);
        if (j0 == dTile || j0 == pTile) {
            // rare path (at most 1 of 128 tiles): mask diagonal / positive entries
#pragma unroll
            for (int r = 0; r < 8; ++r) {
                float s   = S[srow[r] + jMod];
                float wgt = __builtin_fmaf(-s, s, 1.0f);
                float val = acc[r] * K_SCALE * wgt;
                const bool dead = (j == iRow[r]) | (j == jpRow[r]);
                val = dead ? -__builtin_inff() : val;
                float mnew = fmaxf(mx[r], val);               // always finite
                sm[r] = __builtin_fmaf(sm[r], exp2f(mx[r] - mnew), exp2f(val - mnew));
                mx[r] = mnew;
            }
        } else {
            // fast path: no masked entries possible in this tile
#pragma unroll
            for (int r = 0; r < 8; ++r) {
                float s   = S[srow[r] + jMod];
                float wgt = __builtin_fmaf(-s, s, 1.0f);
                float val = acc[r] * K_SCALE * wgt;
                float mnew = fmaxf(mx[r], val);
                sm[r] = __builtin_fmaf(sm[r], exp2f(mx[r] - mnew), exp2f(val - mnew));
                mx[r] = mnew;
            }
        }
    }

    // ---- cross-lane LSE merge within wave via LDS ----
    __syncthreads();
#pragma unroll
    for (int r = 0; r < 8; ++r) {
        redM[w][g * 8 + r][nCol] = mx[r];
        redS[w][g * 8 + r][nCol] = sm[r];
    }
    __syncthreads();

    if (l < 16) {
        const int m = l;
        float M = NEG_BIG;
#pragma unroll
        for (int q = 0; q < 16; ++q) M = fmaxf(M, redM[w][m][q]);
        float T = 0.0f;
#pragma unroll
        for (int q = 0; q < 16; ++q)
            T = __builtin_fmaf(redS[w][m][q], exp2f(redM[w][m][q] - M), T);
        const int n = i0 + m;
        mxW[n * COL_CHUNKS + chunk] = M;
        smW[n * COL_CHUNKS + chunk] = T;
    }
}

// ---------------------------------------------------------------------------
// Kernel 2: merge column-chunk partials + pos into per-row loss, reduce to scalar
// Everything in base-2; single ln2 conversion at the end. Fixed-order -> deterministic.
// ---------------------------------------------------------------------------
__global__ __launch_bounds__(256) void simclr_finalize(
    const float* __restrict__ ws, float* __restrict__ out)
{
    __shared__ float red[256];
    const int t = threadIdx.x;
    const float* posW = ws;
    const float* mxW  = ws + N_SIZE;
    const float* smW  = ws + N_SIZE + N_SIZE * COL_CHUNKS;

    float acc = 0.0f;
    for (int k = 0; k < N_SIZE / 256; ++k) {
        const int n  = t + k * 256;
        const float p2 = posW[n];
        float M = p2;
#pragma unroll
        for (int c = 0; c < COL_CHUNKS; ++c)
            M = fmaxf(M, mxW[n * COL_CHUNKS + c]);
        float T = exp2f(p2 - M);
#pragma unroll
        for (int c = 0; c < COL_CHUNKS; ++c)
            T = __builtin_fmaf(smW[n * COL_CHUNKS + c],
                               exp2f(mxW[n * COL_CHUNKS + c] - M), T);
        acc += (M + log2f(T)) - p2;          // lse2 - pos2 (base-2)
    }
    red[t] = acc;
    __syncthreads();
    for (int s = 128; s > 0; s >>= 1) {
        if (t < s) red[t] += red[t + s];
        __syncthreads();
    }
    if (t == 0) out[0] = red[0] * (LN2_F / (float)N_SIZE);
}

// ---------------------------------------------------------------------------
extern "C" void kernel_launch(void* const* d_in, const int* in_sizes, int n_in,
                              void* d_out, int out_size, void* d_ws, size_t ws_size,
                              hipStream_t stream) {
    const float* hi = (const float*)d_in[0];   // h_i [4096,128] f32
    const float* hj = (const float*)d_in[1];   // h_j [4096,128] f32
    const float* S  = (const float*)d_in[2];   // S   [4096,4096] f32
    float* ws  = (float*)d_ws;                 // needs ~2.4 MB (partials + bf16 h)
    float* out = (float*)d_out;                // scalar f32
    unsigned short* hb = (unsigned short*)(ws + HB_OFF_FLOATS);

    simclr_prep<<<dim3(N_SIZE * D / 8 / 256), dim3(256), 0, stream>>>(hi, hj, hb);
    simclr_pos<<<dim3(N_SIZE / 256), dim3(256), 0, stream>>>(hi, hj, ws);
    simclr_rowpass<<<dim3(256), dim3(256), 0, stream>>>(hb, S, ws);
    simclr_finalize<<<dim3(1), dim3(256), 0, stream>>>(ws, out);
}
